// TESRNN_20925080666192
// MI455X (gfx1250) — matrix operand order repaired and checked by reading, not compiled, not run on hardware
//
#include <hip/hip_runtime.h>

typedef __bf16 v16bf __attribute__((ext_vector_type(16)));
typedef float  v8f   __attribute__((ext_vector_type(8)));

#define IS_   168
#define OS_   24
#define H_    256
#define B_    32
#define T_    1024
#define S_    833      // number of windows / LSTM time steps
#define C0PAD 192      // 168 padded up to multiple of 32

__device__ __forceinline__ unsigned short f2bf(float f) {
  unsigned int u = __float_as_uint(f);
  unsigned int r = u + 0x7FFFu + ((u >> 16) & 1u);   // round-to-nearest-even
  return (unsigned short)(r >> 16);
}
__device__ __forceinline__ float bf2f(unsigned short h) {
  return __uint_as_float(((unsigned int)h) << 16);
}
__device__ __forceinline__ float sigm(float x) { return 1.0f / (1.0f + __expf(-x)); }

// ---------------------------------------------------------------------------
// 1. Exponential-smoothing level scan: one lane per series.
// ---------------------------------------------------------------------------
__global__ void k_levels(const float* __restrict__ train,
                         const float* __restrict__ init_lev_sm,
                         float* __restrict__ levs) {
  int b = threadIdx.x;
  if (b >= B_) return;
  float a = sigm(init_lev_sm[0]);
  const float floorv = 0.1f;              // TAU * MAXIMUM
  float lev = fmaxf(train[b * T_], floorv);
  levs[b * T_] = lev;
  for (int t = 1; t < T_; ++t) {
    lev = fmaxf(a * train[b * T_ + t] + (1.0f - a) * lev, floorv);
    levs[b * T_ + t] = lev;
  }
}

// ---------------------------------------------------------------------------
// 2. Window build: X0 (bf16, [S][32][192], cols >=168 zero) and out_win (f32).
// ---------------------------------------------------------------------------
__global__ void k_windows(const float* __restrict__ train,
                          const float* __restrict__ levs,
                          unsigned short* __restrict__ X0,
                          float* __restrict__ out_win) {
  const int nX = S_ * B_ * C0PAD;
  const int nO = S_ * B_ * OS_;
  for (int i = blockIdx.x * blockDim.x + threadIdx.x; i < nX;
       i += gridDim.x * blockDim.x) {
    int k = i % C0PAD;
    int b = (i / C0PAD) % B_;
    int s = i / (C0PAD * B_);
    float v = 0.f;
    if (k < IS_) v = train[b * T_ + s + k] / levs[b * T_ + s + IS_ - 1];
    X0[i] = f2bf(v);
  }
  for (int i = blockIdx.x * blockDim.x + threadIdx.x; i < nO;
       i += gridDim.x * blockDim.x) {
    int j = i % OS_;
    int b = (i / OS_) % B_;
    int s = i / (OS_ * B_);
    out_win[i] = train[b * T_ + IS_ + s + j] / levs[b * T_ + s + IS_ - 1];
  }
}

// ---------------------------------------------------------------------------
// 3a. Pack LSTM weights [wih ; whh] -> WMMA B-fragment order, bf16.
//     layout: [nt(64)][kc(KC)][lane(32)][half(16)], k_in_chunk = (h&7) + 16*(h>>3) + 8*(L>>4)
// ---------------------------------------------------------------------------
__global__ void k_pack_lstm(const float* __restrict__ wih,
                            const float* __restrict__ whh,
                            const float* __restrict__ bih,
                            const float* __restrict__ bhh,
                            int Cin, int Cpad, int KC,
                            unsigned short* __restrict__ Wdst,
                            float* __restrict__ bdst) {
  int total = 64 * KC * 512;
  for (int i = blockIdx.x * blockDim.x + threadIdx.x; i < total;
       i += gridDim.x * blockDim.x) {
    int h  = i & 15;
    int L  = (i >> 4) & 31;
    int kc = (i >> 9) % KC;
    int nt = i / (KC * 512);
    int n  = nt * 16 + (L & 15);
    int k  = kc * 32 + (h & 7) + 16 * (h >> 3) + 8 * (L >> 4);
    float v = 0.f;
    if (k < Cpad) { if (k < Cin) v = wih[n * Cin + k]; }
    else          { v = whh[n * H_ + (k - Cpad)]; }
    Wdst[i] = f2bf(v);
  }
  for (int i = blockIdx.x * blockDim.x + threadIdx.x; i < 4 * H_;
       i += gridDim.x * blockDim.x)
    bdst[i] = bih[i] + bhh[i];
}

// 3b. Pack dense head weight (rows Nact x 256) -> B fragments, N zero-padded.
__global__ void k_pack_dense(const float* __restrict__ w, int Nact, int Ntiles,
                             unsigned short* __restrict__ dst) {
  int total = Ntiles * 8 * 512;
  for (int i = blockIdx.x * blockDim.x + threadIdx.x; i < total;
       i += gridDim.x * blockDim.x) {
    int h  = i & 15;
    int L  = (i >> 4) & 31;
    int kc = (i >> 9) & 7;
    int nt = i / (8 * 512);
    int n  = nt * 16 + (L & 15);
    int k  = kc * 32 + (h & 7) + 16 * (h >> 3) + 8 * (L >> 4);
    float v = (n < Nact) ? w[n * H_ + k] : 0.f;
    dst[i] = f2bf(v);
  }
}

// ---------------------------------------------------------------------------
// 4. Persistent dilated-LSTM layer. One block = 16 dilated-batch rows.
//    8 waves; wave w owns gate columns [32w, 32w+32) of ALL FOUR gates, so
//    the LSTM pointwise runs fully in registers (c never leaves VGPRs).
// ---------------------------------------------------------------------------
__global__ __launch_bounds__(256) void k_lstm(
    const unsigned short* __restrict__ in, int Cpad, int KC,
    const unsigned short* __restrict__ Wpk, const float* __restrict__ bias,
    int d, int nsteps, unsigned short* __restrict__ out) {
  __shared__ unsigned short Afrag[16 * 512];   // KC<=16, A-fragment order
  __shared__ unsigned short hstate[16 * H_];   // h_{t-1}, bf16

  const int tid   = threadIdx.x;
  const int lane  = tid & 31;
  const int w     = tid >> 5;
  const int q0    = blockIdx.x * 16;  // dilated-batch tile start (mult of 16)
  const int p     = q0 >> 5;          // phase within dilation (orig-time offset)
  const int sbase = q0 & 31;          // series base within the 32 series

  for (int i = tid; i < 16 * H_; i += 256) hstate[i] = 0;

  const int m8  = 8 * (lane >> 4);
  const int nlo = lane & 15;

  // tiles for this wave: acc[2g+j2] <-> nt = g*16 + 2w + j2  (g = i,f,g,o)
  int   ntid[8];
  float bt[8];
#pragma unroll
  for (int g = 0; g < 4; ++g)
#pragma unroll
    for (int j2 = 0; j2 < 2; ++j2) {
      int nt = g * 16 + 2 * w + j2;
      ntid[g * 2 + j2] = nt;
      bt[g * 2 + j2]   = bias[nt * 16 + nlo];
    }

  float creg[16];
#pragma unroll
  for (int i = 0; i < 16; ++i) creg[i] = 0.f;

  __syncthreads();

  for (int step = 0; step < nsteps; ++step) {
    const int t = step * d + p;   // original time index (may be padding)

    // ---- build A fragment [x_t ; h_{t-1}] in fragment order ----
    for (int e = tid; e < KC * 512; e += 256) {
      int h   = e & 15;
      int L   = (e >> 4) & 31;
      int kc  = e >> 9;
      int m   = L & 15;
      int kin = (h & 7) + 16 * (h >> 3) + 8 * (L >> 4);
      int kk  = kc * 32 + kin;
      unsigned short v;
      if (kk < Cpad)
        v = (t < S_) ? in[((size_t)t * B_ + sbase + m) * Cpad + kk]
                     : (unsigned short)0;
      else
        v = hstate[m * H_ + (kk - Cpad)];
      Afrag[e] = v;
    }
    __syncthreads();

    // ---- gates = [x;h] @ W^T via WMMA bf16 ----
    v8f acc[8];
#pragma unroll
    for (int j = 0; j < 8; ++j)
#pragma unroll
      for (int r = 0; r < 8; ++r) acc[j][r] = 0.f;

    for (int kc = 0; kc < KC; ++kc) {
      v16bf a = *reinterpret_cast<const v16bf*>(&Afrag[(kc * 32 + lane) * 16]);
#pragma unroll
      for (int j = 0; j < 8; ++j) {
        v16bf b = *reinterpret_cast<const v16bf*>(
            &Wpk[(((size_t)ntid[j] * KC + kc) * 32 + lane) * 16]);
        acc[j] = __builtin_amdgcn_wmma_f32_16x16x32_bf16(
            false, a, false, b, (short)0, acc[j], false, false);
      }
    }

    // ---- LSTM pointwise, in registers ----
#pragma unroll
    for (int j2 = 0; j2 < 2; ++j2) {
#pragma unroll
      for (int r = 0; r < 8; ++r) {
        float iv = acc[0 + j2][r] + bt[0 + j2];
        float fv = acc[2 + j2][r] + bt[2 + j2];
        float gv = acc[4 + j2][r] + bt[4 + j2];
        float ov = acc[6 + j2][r] + bt[6 + j2];
        float c  = sigm(fv) * creg[j2 * 8 + r] + sigm(iv) * tanhf(gv);
        float hv = sigm(ov) * tanhf(c);
        creg[j2 * 8 + r] = c;
        int jcol = 32 * w + 16 * j2 + nlo;
        int m    = r + m8;
        unsigned short hb = f2bf(hv);
        hstate[m * H_ + jcol] = hb;
        if (t < S_)
          out[((size_t)t * B_ + sbase + m) * H_ + jcol] = hb;
      }
    }
    __syncthreads();
  }
}

// ---------------------------------------------------------------------------
// 5. Head: x = H3 + H1 (residual); pred = tanh(x@nl_w.T + nl_b)@sc_w.T + sc_b
//    One block per 16-row tile (26656 rows total = 1666 tiles exactly).
// ---------------------------------------------------------------------------
__global__ __launch_bounds__(256) void k_head(
    const unsigned short* __restrict__ H3, const unsigned short* __restrict__ H1r,
    const unsigned short* __restrict__ nlpk, const float* __restrict__ nl_b,
    const unsigned short* __restrict__ scpk, const float* __restrict__ sc_b,
    float* __restrict__ pred) {
  __shared__ unsigned short fragA[8 * 512];
  __shared__ unsigned short fragM[8 * 512];
  const int tid  = threadIdx.x;
  const int lane = tid & 31;
  const int w    = tid >> 5;
  const size_t row0 = (size_t)blockIdx.x * 16;

  for (int e = tid; e < 8 * 512; e += 256) {
    int h   = e & 15;
    int L   = (e >> 4) & 31;
    int kc  = e >> 9;
    int m   = L & 15;
    int kin = (h & 7) + 16 * (h >> 3) + 8 * (L >> 4);
    int kk  = kc * 32 + kin;
    size_t idx = (row0 + m) * H_ + kk;
    fragA[e] = f2bf(bf2f(H3[idx]) + bf2f(H1r[idx]));
  }
  __syncthreads();

  const int m8  = 8 * (lane >> 4);
  const int nlo = lane & 15;

  // GEMM1: 16 N-tiles across 8 waves (2 each), tanh, rewrite in A-frag order
#pragma unroll
  for (int j2 = 0; j2 < 2; ++j2) {
    int nt = 2 * w + j2;
    v8f acc;
#pragma unroll
    for (int r = 0; r < 8; ++r) acc[r] = 0.f;
    for (int kc = 0; kc < 8; ++kc) {
      v16bf a = *reinterpret_cast<const v16bf*>(&fragA[(kc * 32 + lane) * 16]);
      v16bf b = *reinterpret_cast<const v16bf*>(
          &nlpk[(((size_t)nt * 8 + kc) * 32 + lane) * 16]);
      acc = __builtin_amdgcn_wmma_f32_16x16x32_bf16(
          false, a, false, b, (short)0, acc, false, false);
    }
    int col  = nt * 16 + nlo;
    float bb = nl_b[col];
    int kc2  = col >> 5;
    int kin2 = col & 31;
    int b2   = (kin2 >> 3) & 1;
    int h2   = (kin2 & 7) + 8 * (kin2 >> 4);
#pragma unroll
    for (int r = 0; r < 8; ++r) {
      float v = tanhf(acc[r] + bb);
      int m  = r + m8;
      int L2 = (m & 15) + 16 * b2;
      fragM[(kc2 * 32 + L2) * 16 + h2] = f2bf(v);
    }
  }
  __syncthreads();

  // GEMM2: 256 -> 24 (padded to 32): waves 0,1 each one N-tile
  if (w < 2) {
    v8f acc;
#pragma unroll
    for (int r = 0; r < 8; ++r) acc[r] = 0.f;
    for (int kc = 0; kc < 8; ++kc) {
      v16bf a = *reinterpret_cast<const v16bf*>(&fragM[(kc * 32 + lane) * 16]);
      v16bf b = *reinterpret_cast<const v16bf*>(
          &scpk[(((size_t)w * 8 + kc) * 32 + lane) * 16]);
      acc = __builtin_amdgcn_wmma_f32_16x16x32_bf16(
          false, a, false, b, (short)0, acc, false, false);
    }
    int n = w * 16 + nlo;
    if (n < OS_) {
      float bb = sc_b[n];
#pragma unroll
      for (int r = 0; r < 8; ++r) {
        size_t row = row0 + r + m8;
        pred[row * OS_ + n] = acc[r] + bb;
      }
    }
  }
}

// ---------------------------------------------------------------------------
extern "C" void kernel_launch(void* const* d_in, const int* in_sizes, int n_in,
                              void* d_out, int out_size, void* d_ws, size_t ws_size,
                              hipStream_t stream) {
  const float* train       = (const float*)d_in[0];
  const float* init_lev_sm = (const float*)d_in[4];
  const float* wih[4] = {(const float*)d_in[5],  (const float*)d_in[9],
                         (const float*)d_in[13], (const float*)d_in[17]};
  const float* whh[4] = {(const float*)d_in[6],  (const float*)d_in[10],
                         (const float*)d_in[14], (const float*)d_in[18]};
  const float* bih[4] = {(const float*)d_in[7],  (const float*)d_in[11],
                         (const float*)d_in[15], (const float*)d_in[19]};
  const float* bhh[4] = {(const float*)d_in[8],  (const float*)d_in[12],
                         (const float*)d_in[16], (const float*)d_in[20]};
  const float* nl_w = (const float*)d_in[21];
  const float* nl_b = (const float*)d_in[22];
  const float* sc_w = (const float*)d_in[23];
  const float* sc_b = (const float*)d_in[24];

  float* pred    = (float*)d_out;                       // 833*32*24
  float* out_win = (float*)d_out + (size_t)S_ * B_ * OS_;

  // ---- workspace layout (bytes) ----
  char* ws = (char*)d_ws;
  size_t off = 0;
  auto alloc = [&](size_t bytes) {
    char* p = ws + off;
    off += (bytes + 255) & ~(size_t)255;
    return p;
  };
  float*          levs = (float*)alloc((size_t)B_ * T_ * 4);
  unsigned short* X0   = (unsigned short*)alloc((size_t)S_ * B_ * C0PAD * 2);
  unsigned short* HA   = (unsigned short*)alloc((size_t)S_ * B_ * H_ * 2);
  unsigned short* HB   = (unsigned short*)alloc((size_t)S_ * B_ * H_ * 2);
  unsigned short* HC   = (unsigned short*)alloc((size_t)S_ * B_ * H_ * 2);
  unsigned short* W0   = (unsigned short*)alloc((size_t)64 * 14 * 512 * 2);
  unsigned short* W1   = (unsigned short*)alloc((size_t)64 * 16 * 512 * 2);
  unsigned short* W2   = (unsigned short*)alloc((size_t)64 * 16 * 512 * 2);
  unsigned short* W3   = (unsigned short*)alloc((size_t)64 * 16 * 512 * 2);
  unsigned short* nlpk = (unsigned short*)alloc((size_t)16 * 8 * 512 * 2);
  unsigned short* scpk = (unsigned short*)alloc((size_t)2 * 8 * 512 * 2);
  float* b0 = (float*)alloc(4096);
  float* b1 = (float*)alloc(4096);
  float* b2 = (float*)alloc(4096);
  float* b3 = (float*)alloc(4096);
  (void)ws_size; (void)in_sizes; (void)n_in; (void)out_size;

  // 1. levels scan
  k_levels<<<1, 32, 0, stream>>>(train, init_lev_sm, levs);

  // 2. windows (also writes out_win half of d_out)
  k_windows<<<2048, 256, 0, stream>>>(train, levs, X0, out_win);

  // 3. weight packing (independent of levels; stream order is fine)
  k_pack_lstm<<<1792, 256, 0, stream>>>(wih[0], whh[0], bih[0], bhh[0],
                                        IS_, C0PAD, 14, W0, b0);
  k_pack_lstm<<<2048, 256, 0, stream>>>(wih[1], whh[1], bih[1], bhh[1],
                                        H_, H_, 16, W1, b1);
  k_pack_lstm<<<2048, 256, 0, stream>>>(wih[2], whh[2], bih[2], bhh[2],
                                        H_, H_, 16, W2, b2);
  k_pack_lstm<<<2048, 256, 0, stream>>>(wih[3], whh[3], bih[3], bhh[3],
                                        H_, H_, 16, W3, b3);
  k_pack_dense<<<256, 256, 0, stream>>>(nl_w, H_, 16, nlpk);
  k_pack_dense<<<32, 256, 0, stream>>>(sc_w, OS_, 2, scpk);

  // 4. dilated LSTM stack: d=1,4,16,64 ; blocks = d*32/16
  k_lstm<<<2,   256, 0, stream>>>(X0, C0PAD, 14, W0, b0, 1,  833, HA);
  k_lstm<<<8,   256, 0, stream>>>(HA, H_,    16, W1, b1, 4,  209, HB);
  k_lstm<<<32,  256, 0, stream>>>(HB, H_,    16, W2, b2, 16, 53,  HA);
  k_lstm<<<128, 256, 0, stream>>>(HA, H_,    16, W3, b3, 64, 14,  HC);

  // 5. head: residual + tanh dense + projection
  k_head<<<1666, 256, 0, stream>>>(HC, HB, nlpk, nl_b, scpk, sc_b, pred);
}
